// TensorProduct_17145509446003
// MI455X (gfx1250) — matrix operand order repaired and checked by reading, not compiled, not run on hardware
//
#include <hip/hip_runtime.h>
#include <hip/hip_bf16.h>

// ---------------------------------------------------------------------------
// e3nn tensor product  (64x0+64x1+64x2) x (1x0+1x1+1x2) -> 19 paths, 5184/row
//
// Strategy: fold W3J with y:   M(y)[i,col] = alpha * sum_j w3j[i,j,k] y[j]
// Then per sample:  Z_l1 (64 x K_l1) = X_l1 (64 x (2l1+1)) @ M_l1(y)
// One wave = one sample; WMMA f32_16x16x32_f16 with hi/lo compensated split
// packed into unused K slots (K 0..4: hi*hi, 8..12: lo*hi, 16..20: hi*lo)
// giving ~fp32 accuracy from the f16 matrix path.  HBM-bound (~755 MB moved),
// so all per-lane addressing is 32-bit off scalar (SGPR) sample bases.
// ---------------------------------------------------------------------------

typedef __attribute__((ext_vector_type(16))) _Float16 v16h;
typedef __attribute__((ext_vector_type(8)))  float    v8f;

#define OUT_DIM 5184
#define X_DIM   576
#define Y_DIM   9

// Folded-W table layout in workspace (floats):
//   l1=0: off 0,   [1][9][9]   = 81
//   l1=1: off 81,  [3][27][9]  = 729
//   l1=2: off 810, [5][45][9]  = 2025   (total 2835 floats)
#define WOFF0 0
#define WOFF1 81
#define WOFF2 810

// Per-column output metadata packed as (base<<4)|stride.  base = chunk base
// within the 5184-float row + k;  stride = 2*l3+1 (per-multiplicity stride).
// Chunk order is the reference (l1,l2,l3) iteration order.
#define MC(b,s) (((b) << 4) | (s))
__device__ __constant__ int META0[9] = {
  MC(0,1), MC(64,3), MC(65,3), MC(66,3),
  MC(256,5), MC(257,5), MC(258,5), MC(259,5), MC(260,5)
};
__device__ __constant__ int META1[27] = {
  MC(576,3), MC(577,3), MC(578,3),
  MC(768,1),
  MC(832,3), MC(833,3), MC(834,3),
  MC(1024,5), MC(1025,5), MC(1026,5), MC(1027,5), MC(1028,5),
  MC(1344,3), MC(1345,3), MC(1346,3),
  MC(1536,5), MC(1537,5), MC(1538,5), MC(1539,5), MC(1540,5),
  MC(1856,7), MC(1857,7), MC(1858,7), MC(1859,7), MC(1860,7), MC(1861,7), MC(1862,7)
};
__device__ __constant__ int META2[45] = {
  MC(2304,5), MC(2305,5), MC(2306,5), MC(2307,5), MC(2308,5),
  MC(2624,3), MC(2625,3), MC(2626,3),
  MC(2816,5), MC(2817,5), MC(2818,5), MC(2819,5), MC(2820,5),
  MC(3136,7), MC(3137,7), MC(3138,7), MC(3139,7), MC(3140,7), MC(3141,7), MC(3142,7),
  MC(3584,1),
  MC(3648,3), MC(3649,3), MC(3650,3),
  MC(3840,5), MC(3841,5), MC(3842,5), MC(3843,5), MC(3844,5),
  MC(4160,7), MC(4161,7), MC(4162,7), MC(4163,7), MC(4164,7), MC(4165,7), MC(4166,7),
  MC(4608,9), MC(4609,9), MC(4610,9), MC(4611,9), MC(4612,9), MC(4613,9),
  MC(4614,9), MC(4615,9), MC(4616,9)
};

// ---------------------------------------------------------------------------
// Init kernel: compute the folded real-basis Wigner-3j table in f64, exactly
// mirroring the reference (Racah CG -> complex 3j -> unitary real basis).
// One lane per path (19 paths).  Runs every launch (deterministic, tiny).
// ---------------------------------------------------------------------------
struct PathDesc { int l1, l2, l3, colstart, yoff, woff, ncol; };
__device__ __constant__ PathDesc PATHS[19] = {
  {0,0,0, 0,0, WOFF0, 9}, {0,1,1, 1,1, WOFF0, 9}, {0,2,2, 4,4, WOFF0, 9},
  {1,0,1, 0,0, WOFF1,27}, {1,1,0, 3,1, WOFF1,27}, {1,1,1, 4,1, WOFF1,27},
  {1,1,2, 7,1, WOFF1,27}, {1,2,1,12,4, WOFF1,27}, {1,2,2,15,4, WOFF1,27},
  {1,2,3,20,4, WOFF1,27},
  {2,0,2, 0,0, WOFF2,45}, {2,1,1, 5,1, WOFF2,45}, {2,1,2, 8,1, WOFF2,45},
  {2,1,3,13,1, WOFF2,45}, {2,2,0,20,4, WOFF2,45}, {2,2,1,21,4, WOFF2,45},
  {2,2,2,24,4, WOFF2,45}, {2,2,3,29,4, WOFF2,45}, {2,2,4,36,4, WOFF2,45},
};

__device__ __forceinline__ double dfact(int n) {
  double r = 1.0;
  for (int i = 2; i <= n; ++i) r *= (double)i;
  return r;
}

__device__ double cg_racah(int j1,int m1,int j2,int m2,int j3,int m3) {
  if (m1 + m2 != m3) return 0.0;
  double pre = (2.0*j3+1.0) * dfact(j3+j1-j2) * dfact(j3-j1+j2) * dfact(j1+j2-j3)
             / dfact(j1+j2+j3+1);
  pre *= dfact(j3+m3)*dfact(j3-m3)*dfact(j1-m1)*dfact(j1+m1)*dfact(j2-m2)*dfact(j2+m2);
  double s = 0.0;
  for (int k = 0; k <= j1+j2-j3; ++k) {
    int d0=j1+j2-j3-k, d1=j1-m1-k, d2=j2+m2-k, d3=j3-j2+m1+k, d4=j3-j1-m2+k;
    if (d0<0 || d1<0 || d2<0 || d3<0 || d4<0) continue;
    double t = 1.0/(dfact(k)*dfact(d0)*dfact(d1)*dfact(d2)*dfact(d3)*dfact(d4));
    s += (k & 1) ? -t : t;
  }
  return sqrt(pre) * s;
}

struct Cplx { double re, im; };

// Q[l] entry: complex->real spherical-harmonic change of basis (row r, col c)
__device__ __forceinline__ Cplx qent(int l, int r, int c) {
  const double s2 = 0.70710678118654752440;
  int rm = r - l, cm = c - l;
  Cplx z; z.re = 0.0; z.im = 0.0;
  if (rm == 0) { if (cm == 0) z.re = 1.0; return z; }
  if (rm > 0) {
    int m = rm;
    if (cm == -m)      z.re = s2;
    else if (cm == m)  z.re = (m & 1) ? -s2 : s2;
  } else {
    int m = -rm;
    if (cm == -m)      z.im = s2;
    else if (cm == m)  z.im = (m & 1) ? s2 : -s2;
  }
  return z;
}

// T[i,j,k] = sum_{a,b,c} Q1[i,a] Q2[j,b] Q3[k,c] C[a,b,c]  (complex)
__device__ void t_entry(int l1,int l2,int l3,int i,int j,int k,
                        double* ore, double* oim) {
  double are = 0.0, aim = 0.0;
  for (int a = 0; a < 2*l1+1; ++a) {
    int m1 = a - l1;
    Cplx q1 = qent(l1, i, a);
    if (q1.re == 0.0 && q1.im == 0.0) continue;
    for (int b = 0; b < 2*l2+1; ++b) {
      int m2 = b - l2;
      int m3 = -(m1 + m2);
      if (m3 < -l3 || m3 > l3) continue;
      Cplx q2 = qent(l2, j, b);
      if (q2.re == 0.0 && q2.im == 0.0) continue;
      Cplx q3 = qent(l3, k, m3 + l3);
      if (q3.re == 0.0 && q3.im == 0.0) continue;
      int p = l1 - l2 - m3;
      double ph = (p & 1) ? -1.0 : 1.0;
      double Cv = ph / sqrt(2.0*l3 + 1.0) * cg_racah(l1, m1, l2, m2, l3, -m3);
      if (Cv == 0.0) continue;
      double r1 = q1.re*q2.re - q1.im*q2.im;
      double i1 = q1.re*q2.im + q1.im*q2.re;
      double r2 = r1*q3.re - i1*q3.im;
      double i2 = r1*q3.im + i1*q3.re;
      are += r2 * Cv;
      aim += i2 * Cv;
    }
  }
  *ore = are; *oim = aim;
}

__global__ void tp_init_w3j(float* __restrict__ Wt) {
  int t = threadIdx.x;
  if (t >= 19) return;
  PathDesc p = PATHS[t];
  int d1 = 2*p.l1+1, d2 = 2*p.l2+1, d3 = 2*p.l3+1;
  // pass 1: choose real vs imaginary part (larger L1 mass)
  double sre = 0.0, sim = 0.0;
  for (int i = 0; i < d1; ++i)
    for (int j = 0; j < d2; ++j)
      for (int k = 0; k < d3; ++k) {
        double re, im;
        t_entry(p.l1, p.l2, p.l3, i, j, k, &re, &im);
        sre += fabs(re); sim += fabs(im);
      }
  bool useRe = (sre >= sim);
  double alpha = sqrt((double)d3);  // component normalization
  // pass 2: write folded table  W[i][colstart+k][jj], jj spans all 9 y comps
  for (int i = 0; i < d1; ++i)
    for (int k = 0; k < d3; ++k)
      for (int jj = 0; jj < 9; ++jj) {
        float v = 0.0f;
        if (jj >= p.yoff && jj < p.yoff + d2) {
          double re, im;
          t_entry(p.l1, p.l2, p.l3, i, jj - p.yoff, k, &re, &im);
          v = (float)(alpha * (useRe ? re : im));
        }
        Wt[p.woff + (i * p.ncol + p.colstart + k) * 9 + jj] = v;
      }
}

// ---------------------------------------------------------------------------
// Main kernel: one wave per sample.  All pointers are per-sample bases
// (scalar / SGPR); every per-lane index is a small 32-bit int.
// ---------------------------------------------------------------------------
template<int KD, int NCOL, int NT>
__device__ __forceinline__ void tp_block(
    const float* __restrict__ xp,   // sample base of this l1 chunk
    const float* __restrict__ Wt,   // folded-W table for this l1
    const float yv[9],
    float* __restrict__ op,         // sample base of output row
    int lane, const int* __restrict__ meta)
{
  const int  colL   = lane & 15;
  const bool hiHalf = (lane < 16);
  const int  rowoff = (lane >> 4) << 3;   // lanes 16..31 hold rows M+8

  // Per-column-tile state, hoisted out of the utile loop.
  v16h b[NT];
  int  cs[NT];      // output stride (2*l3+1)
  int  off0[NT];    // base + rowoff*stride  (32-bit offset from op)
  bool valid[NT];

  #pragma unroll
  for (int t = 0; t < NT; ++t) {
    const int col = t * 16 + colL;
    valid[t] = (col < NCOL);
    int m = 0;
    if (valid[t]) m = meta[col];
    cs[t]   = m & 15;
    off0[t] = (m >> 4) + rowoff * cs[t];

    // B tile = M(y) in WMMA layout.  hi at K=0..KD-1 and duplicated at
    // K=8.. (pairs with x_lo); lo at K=16.. (lower half-wave lanes, elems 8..).
    v16h bb;
    #pragma unroll
    for (int e = 0; e < 16; ++e) bb[e] = (_Float16)0.0f;
    if (valid[t]) {
      #pragma unroll
      for (int h = 0; h < KD; ++h) {
        float s = 0.0f;
        #pragma unroll
        for (int j = 0; j < 9; ++j)
          s += Wt[(h * NCOL + col) * 9 + j] * yv[j];
        _Float16 hi = (_Float16)s;
        if (hiHalf) { bb[h] = hi; bb[8 + h] = (_Float16)(s - (float)hi); }
        else        { bb[h] = hi; }
      }
    }
    b[t] = bb;
  }

  #pragma unroll
  for (int ut = 0; ut < 4; ++ut) {
    // A tile: rows u = ut*16 + (lane&15); both half-waves carry the same row.
    v16h a;
    #pragma unroll
    for (int e = 0; e < 16; ++e) a[e] = (_Float16)0.0f;
    const int u = ut * 16 + colL;
    #pragma unroll
    for (int h = 0; h < KD; ++h) {
      float xv = xp[u * KD + h];
      _Float16 hi = (_Float16)xv;
      if (hiHalf) { a[h] = hi; a[8 + h] = hi; }          // K=0..4 and K=16..20
      else        { a[h] = (_Float16)(xv - (float)hi); } // K=8..12 (lo part)
    }

    #pragma unroll
    for (int t = 0; t < NT; ++t) {
      v8f c = {};
      c = __builtin_amdgcn_wmma_f32_16x16x32_f16(
              /*neg_a=*/false, a, /*neg_b=*/false, b[t],
              /*c_mod=*/(short)0, c, /*reuse_a=*/false, /*reuse_b=*/false);
      if (valid[t]) {
        const int s  = cs[t];
        int off = off0[t] + (ut * 16) * s;   // 32-bit offset from op
        #pragma unroll
        for (int v = 0; v < 8; ++v) {
          op[off] = c[v];
          off += s;
        }
      }
    }
  }
}

__global__ __launch_bounds__(256)
void tp_main(const float* __restrict__ x, const float* __restrict__ y,
             const float* __restrict__ Wt, float* __restrict__ out, int ntot)
{
  const int wid  = (int)((blockIdx.x * blockDim.x + threadIdx.x) >> 5);
  const int lane = (int)(threadIdx.x & 31);
  // wid is wave-uniform: force it scalar so sample bases live in SGPRs and
  // the exit below is a scalar branch (EXEC stays all-1 for WMMA).
  const int ns = __builtin_amdgcn_readfirstlane(wid);
  if (ns >= ntot) return;

  const float* xp = x   + (size_t)ns * X_DIM;
  const float* yp = y   + (size_t)ns * Y_DIM;
  float*       op = out + (size_t)ns * OUT_DIM;

  float yv[9];
  #pragma unroll
  for (int j = 0; j < 9; ++j) yv[j] = yp[j];   // uniform -> SMEM broadcast

  tp_block<1,  9, 1>(xp,       Wt + WOFF0, yv, op, lane, META0);
  tp_block<3, 27, 2>(xp +  64, Wt + WOFF1, yv, op, lane, META1);
  tp_block<5, 45, 3>(xp + 256, Wt + WOFF2, yv, op, lane, META2);
}

// ---------------------------------------------------------------------------
extern "C" void kernel_launch(void* const* d_in, const int* in_sizes, int n_in,
                              void* d_out, int out_size, void* d_ws, size_t ws_size,
                              hipStream_t stream) {
  const float* x  = (const float*)d_in[0];
  const float* y  = (const float*)d_in[1];
  float* out      = (float*)d_out;
  float* Wt       = (float*)d_ws;    // 2835 floats used

  int ntot = in_sizes[0] / X_DIM;    // N samples

  tp_init_w3j<<<1, 32, 0, stream>>>(Wt);

  int waves_per_block = 256 / 32;
  int blocks = (ntot + waves_per_block - 1) / waves_per_block;
  tp_main<<<blocks, 256, 0, stream>>>(x, y, Wt, out, ntot);
}
